// GATLSTM_31782757990559
// MI455X (gfx1250) — compile-verified, matrix operand order checked by reading, not run on hardware
//
#include <hip/hip_runtime.h>
#include <hip/hip_bf16.h>

typedef __attribute__((ext_vector_type(16))) _Float16 v16h;
typedef __attribute__((ext_vector_type(8)))  float    v8f;

#define Bsz 32
#define Tn  64
#define FIN 256
#define Nn  256

// ---------------- WMMA fragment helper (ISA 7.12.2 16-bit layout) -----------
// lane<16:  idx=lane,    halves 0..7 -> K 0..7,   halves 8..15 -> K 16..23
// lane>=16: idx=lane-16, halves 0..7 -> K 8..15,  halves 8..15 -> K 24..31
// Source must be K-contiguous f16: per lane this is two 16-byte loads.
__device__ __forceinline__ v16h frag_k_contig(const _Float16* p, int ld) {
  int lane = threadIdx.x & 31;
  int idx  = lane & 15;
  int kb   = (lane >> 4) << 3;
  const _Float16* q = p + (size_t)idx * ld + kb;
  v16h v;
#pragma unroll
  for (int j = 0; j < 8; ++j) {
    v[j]     = q[j];
    v[j + 8] = q[16 + j];
  }
  return v;
}

__device__ __forceinline__ v8f wmma16(v16h a, v16h b, v8f c) {
  return __builtin_amdgcn_wmma_f32_16x16x32_f16(false, a, false, b, (short)0, c,
                                                false, false);
}

__device__ __forceinline__ float sigm_(float x) { return 1.f / (1.f + __expf(-x)); }
__device__ __forceinline__ float elu_(float x)  { return x > 0.f ? x : (__expf(x) - 1.f); }

// ---------------- f32 -> f16 convert / transpose-convert --------------------
__global__ void cvt_kernel(const float* __restrict__ in, _Float16* __restrict__ out,
                           int n) {
  int i = blockIdx.x * blockDim.x + threadIdx.x;
  if (i < n) out[i] = (_Float16)in[i];
}

// out[(h*O+o)*K + k] = in[(h*K+k)*O + o]   (K-major f16 weights for WMMA B)
__global__ void tconv_kernel(const float* __restrict__ in, _Float16* __restrict__ out,
                             int H, int K, int O) {
  int i = blockIdx.x * blockDim.x + threadIdx.x;
  if (i >= H * K * O) return;
  int h = i / (K * O);
  int r = i % (K * O);
  int o = r / K, k = r % K;
  out[i] = (_Float16)in[((size_t)h * K + k) * O + o];
}

// ---------------- node features + node_fc + LN(64) -> f16 -------------------
__global__ void node_kernel(const float* __restrict__ x, const float* fcw,
                            const float* fcb, const float* lg, const float* lb,
                            _Float16* __restrict__ h0h) {
  int idx = blockIdx.x * blockDim.x + threadIdx.x;           // b*256+n
  if (idx >= Bsz * Nn) return;
  int b = idx >> 8, n = idx & 255;
  const float* xb = x + (size_t)b * Tn * FIN + n;
  float latest = xb[63 * FIN];
  float s = 0.f, ss = 0.f;
  for (int t = 0; t < Tn; ++t) { float v = xb[t * FIN]; s += v; ss += v * v; }
  float mean = s / 64.f;
  float var  = (ss - 64.f * mean * mean) / 63.f;
  float sd   = fmaxf(sqrtf(fmaxf(var, 0.f)), 1e-8f);
  float z    = (latest - mean) / sd;
  float slope = latest - xb[61 * FIN];
  float s5 = 0.f, ss5 = 0.f;
  for (int t = 59; t < 64; ++t) { float v = xb[t * FIN]; s5 += v; ss5 += v * v; }
  float m5 = s5 / 5.f;
  float vol5 = sqrtf(fmaxf((ss5 - 5.f * m5 * m5) / 4.f, 0.f));
  float nf[4] = {latest, z, slope, vol5};
  float f[64];
  float fm = 0.f;
  for (int o = 0; o < 64; ++o) {
    float v = fcb[o];
#pragma unroll
    for (int i = 0; i < 4; ++i) v += nf[i] * fcw[i * 64 + o];
    v = fmaxf(v, 0.f);
    f[o] = v; fm += v;
  }
  fm /= 64.f;
  float fv = 0.f;
  for (int o = 0; o < 64; ++o) { float d = f[o] - fm; fv += d * d; }
  float inv = rsqrtf(fv / 64.f + 1e-5f);
  _Float16* out = h0h + (size_t)idx * 64;
  for (int o = 0; o < 64; ++o) out[o] = (_Float16)((f[o] - fm) * inv * lg[o] + lb[o]);
}

// ---------------- GAT1: Wh = h0 @ W[h] --------------------------------------
__global__ void gat1_wh_kernel(const _Float16* __restrict__ h0h,
                               const _Float16* __restrict__ W1T,
                               float* __restrict__ Wh,
                               _Float16* __restrict__ WhT) {
  int gw = (blockIdx.x * blockDim.x + threadIdx.x) >> 5;
  int ot = gw & 3, nt = (gw >> 2) & 15, h = (gw >> 6) & 7, b = gw >> 9;
  const _Float16* A  = h0h + (size_t)(b * 256 + nt * 16) * 64;
  const _Float16* Bm = W1T + ((size_t)h * 64 + ot * 16) * 64;
  v8f acc = {};
#pragma unroll
  for (int kk = 0; kk < 64; kk += 32)
    acc = wmma16(frag_k_contig(A + kk, 64), frag_k_contig(Bm + kk, 64), acc);
  int lane = threadIdx.x & 31, col = lane & 15, rb = (lane >> 4) * 8;
  float*    D  = Wh  + ((size_t)(b * 8 + h) * 256 + nt * 16) * 64 + ot * 16 + col;
  _Float16* DT = WhT + ((size_t)(b * 8 + h) * 64 + ot * 16 + col) * 256 + nt * 16;
#pragma unroll
  for (int r = 0; r < 8; ++r) {
    D[(size_t)(rb + r) * 64] = acc[r];
    DT[rb + r] = (_Float16)acc[r];
  }
}

// ---------------- attention scores s1/s2 ------------------------------------
__global__ void sdot_kernel(const float* __restrict__ Wh, const float* a1,
                            const float* a2, float* s1, float* s2, int H, int O,
                            int total) {
  int idx = blockIdx.x * blockDim.x + threadIdx.x;   // (b*H+h)*256+n
  if (idx >= total) return;
  int h = (idx >> 8) % H;
  const float* w = Wh + (size_t)idx * O;
  float r1 = 0.f, r2 = 0.f;
  for (int o = 0; o < O; ++o) { float x = w[o]; r1 += x * a1[h * O + o]; r2 += x * a2[h * O + o]; }
  s1[idx] = r1; s2[idx] = r2;
}

// ---------------- masked leaky-relu softmax; optional head-mean output ------
__global__ void att_softmax_kernel(const float* __restrict__ s1,
                                   const float* __restrict__ s2,
                                   const int* __restrict__ adj,
                                   _Float16* __restrict__ att,
                                   float* mean_out, int H) {
  int i = blockIdx.x & 255, b = blockIdx.x >> 8;
  int j = threadIdx.x;                                   // 256 threads
  __shared__ float red[256];
  float acc = 0.f;
  int ok = adj[i * 256 + j] > 0;
  for (int h = 0; h < H; ++h) {
    float v = s1[((size_t)b * H + h) * 256 + i] + s2[((size_t)b * H + h) * 256 + j];
    v = v > 0.f ? v : 0.2f * v;
    if (!ok) v = -9.0e15f;
    red[j] = v; __syncthreads();
    for (int st = 128; st > 0; st >>= 1) { if (j < st) red[j] = fmaxf(red[j], red[j + st]); __syncthreads(); }
    float mx = red[0]; __syncthreads();
    float p = __expf(v - mx);
    red[j] = p; __syncthreads();
    for (int st = 128; st > 0; st >>= 1) { if (j < st) red[j] += red[j + st]; __syncthreads(); }
    float sm = red[0]; __syncthreads();
    float a = p / sm;
    att[(((size_t)b * H + h) * 256 + i) * 256 + j] = (_Float16)a;
    acc += a;
  }
  if (mean_out) mean_out[((size_t)b * 256 + i) * 256 + j] = acc / (float)H;
}

// ---------------- GAT1: hp = elu(att @ Wh), concat-scatter into g1 ----------
__global__ void hp1_kernel(const _Float16* __restrict__ att,
                           const _Float16* __restrict__ WhT,
                           float* __restrict__ g1) {
  int gw = (blockIdx.x * blockDim.x + threadIdx.x) >> 5;
  int ot = gw & 3, nt = (gw >> 2) & 15, h = (gw >> 6) & 7, b = gw >> 9;
  const _Float16* A  = att + ((size_t)(b * 8 + h) * 256 + nt * 16) * 256;
  const _Float16* Bm = WhT + ((size_t)(b * 8 + h) * 64 + ot * 16) * 256;
  v8f acc = {};
#pragma unroll 2
  for (int kk = 0; kk < 256; kk += 32)
    acc = wmma16(frag_k_contig(A + kk, 256), frag_k_contig(Bm + kk, 256), acc);
  int lane = threadIdx.x & 31, col = lane & 15, rb = (lane >> 4) * 8;
#pragma unroll
  for (int r = 0; r < 8; ++r)
    g1[(size_t)(b * 256 + nt * 16 + rb + r) * 512 + h * 64 + ot * 16 + col] = elu_(acc[r]);
}

// ---------------- GAT2: Wh2 = g1 @ W2[h] ------------------------------------
__global__ void wh2_kernel(const _Float16* __restrict__ g1h,
                           const _Float16* __restrict__ W2T,
                           float* __restrict__ Wh2,
                           _Float16* __restrict__ Wh2T) {
  int gw = (blockIdx.x * blockDim.x + threadIdx.x) >> 5;
  int ot = gw & 7, nt = (gw >> 3) & 15, h = (gw >> 7) & 3, b = gw >> 9;
  const _Float16* A  = g1h + (size_t)(b * 256 + nt * 16) * 512;
  const _Float16* Bm = W2T + ((size_t)h * 128 + ot * 16) * 512;
  v8f acc = {};
#pragma unroll 2
  for (int kk = 0; kk < 512; kk += 32)
    acc = wmma16(frag_k_contig(A + kk, 512), frag_k_contig(Bm + kk, 512), acc);
  int lane = threadIdx.x & 31, col = lane & 15, rb = (lane >> 4) * 8;
  float*    D  = Wh2  + ((size_t)(b * 4 + h) * 256 + nt * 16) * 128 + ot * 16 + col;
  _Float16* DT = Wh2T + ((size_t)(b * 4 + h) * 128 + ot * 16 + col) * 256 + nt * 16;
#pragma unroll
  for (int r = 0; r < 8; ++r) {
    D[(size_t)(rb + r) * 128] = acc[r];
    DT[rb + r] = (_Float16)acc[r];
  }
}

// ---------------- GAT2: hp2 = elu(att2 @ Wh2) -------------------------------
__global__ void hp2_kernel(const _Float16* __restrict__ att,
                           const _Float16* __restrict__ Wh2T,
                           float* __restrict__ hp2) {
  int gw = (blockIdx.x * blockDim.x + threadIdx.x) >> 5;
  int ot = gw & 7, nt = (gw >> 3) & 15, h = (gw >> 7) & 3, b = gw >> 9;
  const _Float16* A  = att + ((size_t)(b * 4 + h) * 256 + nt * 16) * 256;
  const _Float16* Bm = Wh2T + ((size_t)(b * 4 + h) * 128 + ot * 16) * 256;
  v8f acc = {};
#pragma unroll 2
  for (int kk = 0; kk < 256; kk += 32)
    acc = wmma16(frag_k_contig(A + kk, 256), frag_k_contig(Bm + kk, 256), acc);
  int lane = threadIdx.x & 31, col = lane & 15, rb = (lane >> 4) * 8;
  float* D = hp2 + ((size_t)(b * 4 + h) * 256 + nt * 16) * 128 + ot * 16 + col;
#pragma unroll
  for (int r = 0; r < 8; ++r) D[(size_t)(rb + r) * 128] = elu_(acc[r]);
}

// ---------------- mean over 4 heads -----------------------------------------
__global__ void g2mean_kernel(const float* __restrict__ hp2, float* __restrict__ g2) {
  int idx = blockIdx.x * blockDim.x + threadIdx.x;
  if (idx >= Bsz * Nn * 128) return;
  int b = idx >> 15, q = idx & 32767;
  float s = 0.f;
#pragma unroll
  for (int h = 0; h < 4; ++h) s += hp2[((size_t)b * 4 + h) * 32768 + q];
  g2[idx] = s * 0.25f;
}

// ---------------- generic LayerNorm (optional relu, optional f16 mirror) ----
__global__ void ln_kernel(const float* __restrict__ in, float* __restrict__ out,
                          const float* g, const float* bta, int L,
                          int in_stride, int out_stride, int do_relu,
                          _Float16* outh) {
  int row = blockIdx.x;
  const float* x = in + (size_t)row * in_stride;
  float* y = out + (size_t)row * out_stride;
  __shared__ float red[128];
  int tid = threadIdx.x;
  float s = 0.f;
  for (int i = tid; i < L; i += 128) s += x[i];
  red[tid] = s; __syncthreads();
  for (int st = 64; st > 0; st >>= 1) { if (tid < st) red[tid] += red[tid + st]; __syncthreads(); }
  float mean = red[0] / (float)L; __syncthreads();
  float v = 0.f;
  for (int i = tid; i < L; i += 128) { float d = x[i] - mean; v += d * d; }
  red[tid] = v; __syncthreads();
  for (int st = 64; st > 0; st >>= 1) { if (tid < st) red[tid] += red[tid + st]; __syncthreads(); }
  float inv = rsqrtf(red[0] / (float)L + 1e-5f); __syncthreads();
  for (int i = tid; i < L; i += 128) {
    float r = (x[i] - mean) * inv * g[i] + bta[i];
    if (do_relu) r = fmaxf(r, 0.f);
    y[i] = r;
    if (outh) outh[(size_t)row * out_stride + i] = (_Float16)r;
  }
}

// ---------------- persistent LSTM layer: 1 block, WMMA gates ----------------
__global__ __launch_bounds__(1024) void lstm_kernel(
    const _Float16* __restrict__ x, int Kin, const _Float16* __restrict__ Wih,
    const _Float16* __restrict__ Whh, const float* __restrict__ bih,
    const float* __restrict__ bhh, float* outf, _Float16* outh) {
  __shared__ _Float16 h_lds[32 * 128];   // 8 KB
  __shared__ float    c_lds[32 * 128];   // 16 KB
  __shared__ _Float16 g_lds[32 * 512];   // 32 KB
  int tid = threadIdx.x;
  for (int i = tid; i < 32 * 128; i += 1024) { h_lds[i] = (_Float16)0.f; c_lds[i] = 0.f; }
  __syncthreads();
  int lane = tid & 31, wv = tid >> 5;
  int col = lane & 15, rb = (lane >> 4) * 8;
  for (int t = 0; t < Tn; ++t) {
#pragma unroll
    for (int it = 0; it < 2; ++it) {
      int tile = wv + it * 32;                // 64 tiles: 2 M-tiles x 32 N-tiles
      int mt = tile & 1, nt = tile >> 1;
      v8f acc = {};
      const _Float16* Ax = x + ((size_t)(mt * 16) * Tn + t) * Kin;
      const _Float16* Bx = Wih + (size_t)(nt * 16) * Kin;
      for (int kk = 0; kk < Kin; kk += 32)
        acc = wmma16(frag_k_contig(Ax + kk, Tn * Kin), frag_k_contig(Bx + kk, Kin), acc);
      const _Float16* Ah = h_lds + (size_t)(mt * 16) * 128;
      const _Float16* Bh = Whh + (size_t)(nt * 16) * 128;
#pragma unroll
      for (int kk = 0; kk < 128; kk += 32)
        acc = wmma16(frag_k_contig(Ah + kk, 128), frag_k_contig(Bh + kk, 128), acc);
      int gc = nt * 16 + col;
      float bias = bih[gc] + bhh[gc];
#pragma unroll
      for (int r = 0; r < 8; ++r)
        g_lds[(size_t)(mt * 16 + rb + r) * 512 + gc] = (_Float16)(acc[r] + bias);
    }
    __syncthreads();
    for (int i = tid; i < 32 * 128; i += 1024) {
      int bb = i >> 7, j = i & 127;
      float gi = (float)g_lds[bb * 512 + j];
      float gf = (float)g_lds[bb * 512 + 128 + j];
      float gg = (float)g_lds[bb * 512 + 256 + j];
      float go = (float)g_lds[bb * 512 + 384 + j];
      float c = c_lds[i];
      c = sigm_(gf) * c + sigm_(gi) * tanhf(gg);
      float hh = sigm_(go) * tanhf(c);
      c_lds[i] = c;
      h_lds[i] = (_Float16)hh;
      if (outf) outf[((size_t)bb * Tn + t) * 128 + j] = hh;
      if (outh) outh[((size_t)bb * Tn + t) * 128 + j] = (_Float16)hh;
    }
    __syncthreads();
  }
}

// ---------------- fused = [gat_flat | lstm_last] @ fcs_w + b ----------------
__global__ void fcs_kernel(const float* __restrict__ g2flat,
                           const float* __restrict__ last,
                           const float* __restrict__ W, const float* bias,
                           float* __restrict__ out) {
  int wvg = (blockIdx.x * blockDim.x + threadIdx.x) >> 5;   // 4096 waves
  int lane = threadIdx.x & 31;
  int b = wvg >> 7, o = wvg & 127;
  float s = 0.f;
  for (int k = lane; k < 32768; k += 32) {
    if (((k >> 5) & 63) == 0 && k + 8192 < 32768)
      __builtin_prefetch(&W[(size_t)(k + 8192) * 128 + o], 0, 0);  // global_prefetch_b8
    s += g2flat[(size_t)b * 32768 + k] * W[(size_t)k * 128 + o];
  }
  for (int k = 32768 + lane; k < 32896; k += 32)
    s += last[b * 128 + (k - 32768)] * W[(size_t)k * 128 + o];
  for (int m = 16; m > 0; m >>= 1) s += __shfl_xor(s, m, 32);
  if (lane == 0) out[b * 128 + o] = s + bias[o];
}

// ---------------- dir / ret / vol heads -------------------------------------
__global__ void heads_kernel(const float* __restrict__ sh, const float* dw,
                             const float* db, const float* rw, const float* rb_,
                             const float* vw, const float* vb,
                             float* __restrict__ out) {
  int tid = threadIdx.x;
  if (tid >= 160) return;
  int b = tid / 5, k = tid % 5;
  const float* s = sh + b * 128;
  float acc = 0.f;
  if (k < 3) {
    for (int j = 0; j < 128; ++j) acc += s[j] * dw[j * 3 + k];
    out[b * 3 + k] = acc + db[k];
  } else if (k == 3) {
    for (int j = 0; j < 128; ++j) acc += s[j] * rw[j];
    out[96 + b] = acc + rb_[0];
  } else {
    for (int j = 0; j < 128; ++j) acc += s[j] * vw[j];
    out[128 + b] = acc + vb[0];
  }
}

// ---------------------------------------------------------------------------
extern "C" void kernel_launch(void* const* d_in, const int* in_sizes, int n_in,
                              void* d_out, int out_size, void* d_ws, size_t ws_size,
                              hipStream_t stream) {
  const float* x_seq    = (const float*)d_in[0];
  const float* node_fcw = (const float*)d_in[1];
  const float* node_fcb = (const float*)d_in[2];
  const float* node_lg  = (const float*)d_in[3];
  const float* node_lb  = (const float*)d_in[4];
  const float* gat1_W   = (const float*)d_in[5];
  const float* gat1_a1  = (const float*)d_in[6];
  const float* gat1_a2  = (const float*)d_in[7];
  const float* gat1_lg  = (const float*)d_in[8];
  const float* gat1_lb  = (const float*)d_in[9];
  const float* gat2_W   = (const float*)d_in[10];
  const float* gat2_a1  = (const float*)d_in[11];
  const float* gat2_a2  = (const float*)d_in[12];
  const float* gat2_lg  = (const float*)d_in[13];
  const float* gat2_lb  = (const float*)d_in[14];
  const float* Wih0     = (const float*)d_in[15];
  const float* Whh0     = (const float*)d_in[16];
  const float* bih0     = (const float*)d_in[17];
  const float* bhh0     = (const float*)d_in[18];
  const float* Wih1     = (const float*)d_in[19];
  const float* Whh1     = (const float*)d_in[20];
  const float* bih1     = (const float*)d_in[21];
  const float* bhh1     = (const float*)d_in[22];
  const float* lstm_lg  = (const float*)d_in[23];
  const float* lstm_lb  = (const float*)d_in[24];
  const float* fcs_w    = (const float*)d_in[25];
  const float* fcs_b    = (const float*)d_in[26];
  const float* fcs_lg   = (const float*)d_in[27];
  const float* fcs_lb   = (const float*)d_in[28];
  const float* dir_w    = (const float*)d_in[29];
  const float* dir_b    = (const float*)d_in[30];
  const float* ret_w    = (const float*)d_in[31];
  const float* ret_b    = (const float*)d_in[32];
  const float* vol_w    = (const float*)d_in[33];
  const float* vol_b    = (const float*)d_in[34];
  const int*   adj      = (const int*)d_in[35];
  float* out = (float*)d_out;

  // ---- workspace carve ----
  char* wp = (char*)d_ws;
  auto carve = [&](size_t bytes) -> void* {
    void* p = (void*)wp;
    wp += (bytes + 255) & ~(size_t)255;
    return p;
  };
  _Float16* h0h   = (_Float16*)carve((size_t)Bsz * Nn * 64 * 2);        // 1 MB
  _Float16* W1T   = (_Float16*)carve((size_t)8 * 64 * 64 * 2);
  float*    Wh1   = (float*)   carve((size_t)Bsz * 8 * Nn * 64 * 4);    // 16 MB
  _Float16* Wh1T  = (_Float16*)carve((size_t)Bsz * 8 * Nn * 64 * 2);    // 8 MB
  float*    s11   = (float*)   carve((size_t)Bsz * 8 * Nn * 4);
  float*    s21   = (float*)   carve((size_t)Bsz * 8 * Nn * 4);
  _Float16* att1  = (_Float16*)carve((size_t)Bsz * 8 * Nn * Nn * 2);    // 32 MB
  float*    g1    = (float*)   carve((size_t)Bsz * Nn * 512 * 4);       // 16 MB
  _Float16* g1h   = (_Float16*)carve((size_t)Bsz * Nn * 512 * 2);       // 8 MB
  _Float16* W2T   = (_Float16*)carve((size_t)4 * 512 * 128 * 2);
  float*    Wh2   = (float*)   carve((size_t)Bsz * 4 * Nn * 128 * 4);   // 16 MB
  _Float16* Wh2T  = (_Float16*)carve((size_t)Bsz * 4 * Nn * 128 * 2);   // 8 MB
  float*    s12   = (float*)   carve((size_t)Bsz * 4 * Nn * 4);
  float*    s22   = (float*)   carve((size_t)Bsz * 4 * Nn * 4);
  _Float16* att2  = (_Float16*)carve((size_t)Bsz * 4 * Nn * Nn * 2);    // 16 MB
  float*    hp2   = (float*)   carve((size_t)Bsz * 4 * Nn * 128 * 4);   // 16 MB
  float*    g2    = (float*)   carve((size_t)Bsz * Nn * 128 * 4);       // 4 MB
  _Float16* x16   = (_Float16*)carve((size_t)Bsz * Tn * FIN * 2);       // 1 MB
  _Float16* Wih0h = (_Float16*)carve((size_t)512 * 256 * 2);
  _Float16* Whh0h = (_Float16*)carve((size_t)512 * 128 * 2);
  _Float16* Wih1h = (_Float16*)carve((size_t)512 * 128 * 2);
  _Float16* Whh1h = (_Float16*)carve((size_t)512 * 128 * 2);
  _Float16* h1h   = (_Float16*)carve((size_t)Bsz * Tn * 128 * 2);
  float*    h2seq = (float*)   carve((size_t)Bsz * Tn * 128 * 4);
  float*    last  = (float*)   carve((size_t)Bsz * 128 * 4);
  float*    spre  = (float*)   carve((size_t)Bsz * 128 * 4);
  float*    shrd  = (float*)   carve((size_t)Bsz * 128 * 4);

  float* att_out = out + 160;  // dir(96) + ret(32) + vol(32) then att

  // ---- one-time (per call) f16 weight/input staging ----
  cvt_kernel<<<2048, 256, 0, stream>>>(x_seq, x16, Bsz * Tn * FIN);
  cvt_kernel<<<512, 256, 0, stream>>>(Wih0, Wih0h, 512 * 256);
  cvt_kernel<<<256, 256, 0, stream>>>(Whh0, Whh0h, 512 * 128);
  cvt_kernel<<<256, 256, 0, stream>>>(Wih1, Wih1h, 512 * 128);
  cvt_kernel<<<256, 256, 0, stream>>>(Whh1, Whh1h, 512 * 128);
  tconv_kernel<<<128, 256, 0, stream>>>(gat1_W, W1T, 8, 64, 64);
  tconv_kernel<<<1024, 256, 0, stream>>>(gat2_W, W2T, 4, 512, 128);

  // ---- graph neural path ----
  node_kernel<<<32, 256, 0, stream>>>(x_seq, node_fcw, node_fcb, node_lg, node_lb, h0h);

  gat1_wh_kernel<<<2048, 256, 0, stream>>>(h0h, W1T, Wh1, Wh1T);        // 16384 waves
  sdot_kernel<<<256, 256, 0, stream>>>(Wh1, gat1_a1, gat1_a2, s11, s21, 8, 64, Bsz * 8 * Nn);
  att_softmax_kernel<<<Bsz * Nn, 256, 0, stream>>>(s11, s21, adj, att1, nullptr, 8);
  hp1_kernel<<<2048, 256, 0, stream>>>(att1, Wh1T, g1);
  ln_kernel<<<Bsz * Nn, 128, 0, stream>>>(g1, g1, gat1_lg, gat1_lb, 512, 512, 512, 0, g1h);

  wh2_kernel<<<2048, 256, 0, stream>>>(g1h, W2T, Wh2, Wh2T);
  sdot_kernel<<<128, 256, 0, stream>>>(Wh2, gat2_a1, gat2_a2, s12, s22, 4, 128, Bsz * 4 * Nn);
  att_softmax_kernel<<<Bsz * Nn, 256, 0, stream>>>(s12, s22, adj, att2, att_out, 4);
  hp2_kernel<<<2048, 256, 0, stream>>>(att2, Wh2T, hp2);
  g2mean_kernel<<<4096, 256, 0, stream>>>(hp2, g2);
  ln_kernel<<<Bsz * Nn, 128, 0, stream>>>(g2, g2, gat2_lg, gat2_lb, 128, 128, 128, 0, nullptr);

  // ---- recurrent path (sequential; persistent WMMA blocks) ----
  lstm_kernel<<<1, 1024, 0, stream>>>(x16, FIN, Wih0h, Whh0h, bih0, bhh0, nullptr, h1h);
  lstm_kernel<<<1, 1024, 0, stream>>>(h1h, 128, Wih1h, Whh1h, bih1, bhh1, h2seq, nullptr);
  ln_kernel<<<Bsz, 128, 0, stream>>>(h2seq + 63 * 128, last, lstm_lg, lstm_lb, 128,
                                     Tn * 128, 128, 0, nullptr);

  // ---- fusion head ----
  fcs_kernel<<<512, 256, 0, stream>>>(g2, last, fcs_w, fcs_b, spre);
  ln_kernel<<<Bsz, 128, 0, stream>>>(spre, shrd, fcs_lg, fcs_lb, 128, 128, 128, 1, nullptr);
  heads_kernel<<<1, 192, 0, stream>>>(shrd, dir_w, dir_b, ret_w, ret_b, vol_w, vol_b, out);
}
